// DecoderWithAttention_76974403879145
// MI455X (gfx1250) — matrix operand order repaired and checked by reading, not compiled, not run on hardware
//
#include <hip/hip_runtime.h>
#include <hip/hip_bf16.h>
#include <math.h>

typedef _Float16 half_t;
typedef __attribute__((ext_vector_type(16))) _Float16 v16h;
typedef __attribute__((ext_vector_type(8)))  _Float16 v8h;
typedef __attribute__((ext_vector_type(8)))  float    v8f;

#define Bsz   128
#define Ppix  196
#define Edim  1792
#define Adim  512
#define Ddim  512
#define EMBd  512
#define Vvoc  10000
#define Lcap  52
#define Tstep 51
#define Gdim  2048   /* 4*D */
#define Xdim  2304   /* EMB + E */

// ---------------------------------------------------------------------------
// f16 WMMA GEMM, compile-time 1xNT tile blocking per wave:
//   C[M][N] = act( A[M][K] @ Bt[N][K]^T + bias ) * rowScale
// A row-major f16 (lda halfs), Bt row-major f16 [N][K] (ldb halfs), C f32 (ldc).
// Each wave computes a 16x(16*NT) strip: one A fragment load feeds NT WMMAs.
// NT is a template parameter -> straight-line k-loop, no EXEC games.
// ---------------------------------------------------------------------------
template<int NT>
__global__ void gemm_wmma_nt(const half_t* __restrict__ A, int lda,
                             const half_t* __restrict__ Bt, int ldb,
                             const float* __restrict__ bias,
                             const float* __restrict__ rowScale,
                             float* __restrict__ C, int ldc,
                             int M, int K, int nGroups, int tileNoff,
                             int accumulate, int act)
{
    int wave = blockIdx.x * (blockDim.x >> 5) + (threadIdx.x >> 5);
    int lane = threadIdx.x & 31;
    int tilesM = M >> 4;
    int tM  = wave / nGroups;
    int tNB = wave - tM * nGroups;
    if (tM >= tilesM) return;                 // whole-wave early exit
    int tN0 = tileNoff + tNB * NT;

    int r15    = lane & 15;
    int colSel = lane >> 4;                   // 0 or 1

    const half_t* Arow = A  + (size_t)(tM * 16 + r15) * lda;
    const half_t* Brow = Bt + (size_t)(tN0 * 16 + r15) * ldb;
    const size_t  bstep = (size_t)16 * ldb;   // next N-tile of Bt

    v8f acc[NT] = {};
    for (int k0 = 0; k0 < K; k0 += 32) {
        v16h a;
        // A 16x32 layout: halfs 0..7 = K[kb..kb+7], halfs 8..15 = K[kb+16..kb+23], kb = colSel*8
        ((v8h*)&a)[0] = *(const v8h*)(Arow + k0 + colSel * 8);
        ((v8h*)&a)[1] = *(const v8h*)(Arow + k0 + colSel * 8 + 16);
#pragma unroll
        for (int j = 0; j < NT; ++j) {
            // B 32x16 layout: halfs 0..7 = K[kb..kb+7], halfs 8..15 = K[kb+16..kb+23], kb = colSel*8 ; N = lane&15
            v16h b;
            const half_t* Bp = Brow + (size_t)j * bstep + k0 + colSel * 8;
            ((v8h*)&b)[0] = *(const v8h*)(Bp);
            ((v8h*)&b)[1] = *(const v8h*)(Bp + 16);
            acc[j] = __builtin_amdgcn_wmma_f32_16x16x32_f16(false, a, false, b,
                                                            (short)0, acc[j], false, false);
        }
    }

    // C/D layout: VGPR r -> lanes0-15: (M=r, N=lane); lanes16-31: (M=r+8, N=lane-16)
#pragma unroll
    for (int j = 0; j < NT; ++j) {
        int outN = (tN0 + j) * 16 + r15;
#pragma unroll
        for (int r = 0; r < 8; ++r) {
            int outM = tM * 16 + r + colSel * 8;
            float v = acc[j][r];
            if (bias)       v += bias[outN];
            if (accumulate) v += C[(size_t)outM * ldc + outN];
            if (act == 1)   v  = tanhf(v);
            if (rowScale)   v *= rowScale[outM];
            C[(size_t)outM * ldc + outN] = v;
        }
    }
}

// ---------------------------------------------------------------------------
// Setup kernels
// ---------------------------------------------------------------------------
__global__ void order_kernel(const int* __restrict__ lens, int* __restrict__ order_i,
                             int* __restrict__ declen_i, float* __restrict__ out_declen,
                             float* __restrict__ out_order)
{
    __shared__ int L[Bsz];
    int i = threadIdx.x;
    L[i] = lens[i];
    __syncthreads();
    int li = L[i], rank = 0;
    for (int j = 0; j < Bsz; ++j) {
        int lj = L[j];
        if (lj > li || (lj == li && j < i)) rank++;
    }
    order_i[rank] = i;                        // stable argsort of -lens
    __syncthreads();
    int src = order_i[i];
    int dl  = L[src] - 1;
    declen_i[i]   = dl;
    out_declen[i] = (float)dl;
    out_order[i]  = (float)src;
}

__global__ void caps_kernel(const int* __restrict__ caps, const int* __restrict__ order_i,
                            int* __restrict__ caps_i, float* __restrict__ out_caps)
{
    int idx = blockIdx.x * blockDim.x + threadIdx.x;
    if (idx >= Bsz * Lcap) return;
    int b = idx / Lcap, t = idx % Lcap;
    int v = caps[order_i[b] * Lcap + t];
    caps_i[idx]  = v;
    out_caps[idx] = (float)v;
}

__global__ void gather_enc_kernel(const float* __restrict__ enc, const int* __restrict__ order_i,
                                  half_t* __restrict__ encH)
{
    size_t idx = (size_t)blockIdx.x * blockDim.x + threadIdx.x;
    const size_t PE = (size_t)Ppix * Edim;
    if (idx >= (size_t)Bsz * PE) return;
    int b = (int)(idx / PE);
    size_t r = idx - (size_t)b * PE;
    encH[idx] = (half_t)enc[(size_t)order_i[b] * PE + r];
}

__global__ void mean_kernel(const float* __restrict__ enc, const int* __restrict__ order_i,
                            half_t* __restrict__ meanH)
{
    int idx = blockIdx.x * blockDim.x + threadIdx.x;
    if (idx >= Bsz * Edim) return;
    int b = idx / Edim, e = idx % Edim;
    const float* base = enc + (size_t)order_i[b] * Ppix * Edim + e;
    float s = 0.f;
    for (int p = 0; p < Ppix; ++p) s += base[(size_t)p * Edim];
    meanH[idx] = (half_t)(s * (1.0f / Ppix));
}

__global__ void embs_kernel(const int* __restrict__ caps_i, const float* __restrict__ emb,
                            half_t* __restrict__ embsH)
{
    size_t idx = (size_t)blockIdx.x * blockDim.x + threadIdx.x;
    if (idx >= (size_t)Tstep * Bsz * EMBd) return;
    int t   = (int)(idx / (Bsz * EMBd));
    int rem = (int)(idx % (Bsz * EMBd));
    int b = rem / EMBd, e = rem % EMBd;
    int tok = caps_i[b * Lcap + t];
    embsH[idx] = (half_t)emb[(size_t)tok * EMBd + e];
}

// W[K][N] (f32, row-major) -> Wt[N][K] (f16, row-major)
__global__ void transposeW_kernel(const float* __restrict__ W, half_t* __restrict__ Wt,
                                  int K, int N)
{
    size_t idx = (size_t)blockIdx.x * blockDim.x + threadIdx.x;
    if (idx >= (size_t)K * N) return;
    int n = (int)(idx / K), k = (int)(idx % K);
    Wt[idx] = (half_t)W[(size_t)k * N + n];
}

__global__ void convert_kernel(const float* __restrict__ W, half_t* __restrict__ Wh, size_t n)
{
    size_t idx = (size_t)blockIdx.x * blockDim.x + threadIdx.x;
    if (idx < n) Wh[idx] = (half_t)W[idx];
}

// ---------------------------------------------------------------------------
// Per-step kernels
// ---------------------------------------------------------------------------
__global__ void attn_e_kernel(const float* __restrict__ att1, const float* __restrict__ att2,
                              const float* __restrict__ w_fa, const float* __restrict__ b_fa,
                              float* __restrict__ evec)
{
    int wavesPerBlock = blockDim.x >> 5;
    int w    = blockIdx.x * wavesPerBlock + (threadIdx.x >> 5);
    int lane = threadIdx.x & 31;
    if (w >= Bsz * Ppix) return;
    int b = w / Ppix, p = w % Ppix;
    const float* a1 = att1 + (size_t)w * Adim;
    const float* a2 = att2 + (size_t)b * Adim;
    float s = 0.f;
    for (int k = lane; k < Adim; k += 32) {
        float v = a1[k] + a2[k];
        v = v > 0.f ? v : 0.f;
        s += v * w_fa[k];
    }
    for (int off = 16; off > 0; off >>= 1) s += __shfl_down(s, off, 32);
    if (lane == 0) evec[b * Ppix + p] = s + b_fa[0];
}

__global__ void softmax_kernel(const float* __restrict__ evec, float* __restrict__ alpha,
                               float* __restrict__ out_attn, const int* __restrict__ declen_i,
                               int t)
{
    __shared__ float red[256];
    int b = blockIdx.x, tid = threadIdx.x;
    float m = -1e30f;
    for (int p = tid; p < Ppix; p += blockDim.x) m = fmaxf(m, evec[b * Ppix + p]);
    red[tid] = m; __syncthreads();
    for (int s = 128; s > 0; s >>= 1) { if (tid < s) red[tid] = fmaxf(red[tid], red[tid + s]); __syncthreads(); }
    m = red[0]; __syncthreads();
    float sum = 0.f;
    for (int p = tid; p < Ppix; p += blockDim.x) sum += __expf(evec[b * Ppix + p] - m);
    red[tid] = sum; __syncthreads();
    for (int s = 128; s > 0; s >>= 1) { if (tid < s) red[tid] += red[tid + s]; __syncthreads(); }
    float inv = 1.f / red[0];
    float act = (t < declen_i[b]) ? 1.f : 0.f;
    for (int p = tid; p < Ppix; p += blockDim.x) {
        float a = __expf(evec[b * Ppix + p] - m) * inv;
        alpha[b * Ppix + p] = a;
        out_attn[((size_t)b * Tstep + t) * Ppix + p] = a * act;
    }
}

__global__ void ctx_kernel(const half_t* __restrict__ encH, const float* __restrict__ alpha,
                           float* __restrict__ ctx)
{
    int idx = blockIdx.x * blockDim.x + threadIdx.x;
    if (idx >= Bsz * Edim) return;
    int b = idx / Edim, e = idx % Edim;
    const half_t* base = encH + (size_t)b * Ppix * Edim + e;
    const float* al = alpha + b * Ppix;
    float s = 0.f;
    for (int p = 0; p < Ppix; ++p) s += al[p] * (float)base[(size_t)p * Edim];
    ctx[idx] = s;
}

__global__ void buildx_kernel(const half_t* __restrict__ embsH_t, const float* __restrict__ glin,
                              const float* __restrict__ ctx, half_t* __restrict__ xH)
{
    int idx = blockIdx.x * blockDim.x + threadIdx.x;
    if (idx >= Bsz * Xdim) return;
    int b = idx / Xdim, j = idx % Xdim;
    half_t v;
    if (j < EMBd) {
        v = embsH_t[b * EMBd + j];
    } else {
        int jj = j - EMBd;
        float g = 1.f / (1.f + __expf(-glin[b * Edim + jj]));
        v = (half_t)(g * ctx[b * Edim + jj]);
    }
    xH[idx] = v;
}

__global__ void lstm_kernel(const float* __restrict__ gates, float* __restrict__ h,
                            float* __restrict__ c, half_t* __restrict__ hH,
                            half_t* __restrict__ hnewH, float* __restrict__ amask,
                            const int* __restrict__ declen_i, int t)
{
    int idx = blockIdx.x * blockDim.x + threadIdx.x;
    if (idx >= Bsz * Ddim) return;
    int b = idx / Ddim, d = idx % Ddim;
    const float* g = gates + (size_t)b * Gdim;
    float ig = g[d], fg = g[Ddim + d], gg = g[2 * Ddim + d], og = g[3 * Ddim + d];
    float c_old = c[idx], h_old = h[idx];
    float si = 1.f / (1.f + __expf(-ig));
    float sf = 1.f / (1.f + __expf(-fg));
    float so = 1.f / (1.f + __expf(-og));
    float cn = sf * c_old + si * tanhf(gg);
    float hn = so * tanhf(cn);
    float act = (t < declen_i[b]) ? 1.f : 0.f;
    float h2 = act * hn + (1.f - act) * h_old;
    float c2 = act * cn + (1.f - act) * c_old;
    h[idx] = h2; c[idx] = c2;
    hH[idx] = (half_t)h2;
    hnewH[idx] = (half_t)hn;
    if (d == 0) amask[b] = act;
}

__global__ void f2h_kernel(const float* __restrict__ src, half_t* __restrict__ dst, int n)
{
    int i = blockIdx.x * blockDim.x + threadIdx.x;
    if (i < n) dst[i] = (half_t)src[i];
}

// ---------------------------------------------------------------------------
static inline void launch_gemm(hipStream_t s, const half_t* A, int lda,
                               const half_t* Bt, int ldb, const float* bias,
                               const float* rowScale, float* C, int ldc,
                               int M, int N, int K, int accumulate, int act)
{
    int tilesM = M / 16, tilesN = N / 16;
    int full = tilesN / 4, rem = tilesN % 4;
    if (full) {
        int waves  = tilesM * full;
        int blocks = (waves + 3) / 4;             // 4 waves / block
        gemm_wmma_nt<4><<<blocks, 128, 0, s>>>(A, lda, Bt, ldb, bias, rowScale,
                                               C, ldc, M, K, full, 0, accumulate, act);
    }
    if (rem) {
        int waves  = tilesM;                      // one remainder group per M-tile
        int blocks = (waves + 3) / 4;
        if (rem == 1)
            gemm_wmma_nt<1><<<blocks, 128, 0, s>>>(A, lda, Bt, ldb, bias, rowScale,
                                                   C, ldc, M, K, 1, full * 4, accumulate, act);
        else if (rem == 2)
            gemm_wmma_nt<2><<<blocks, 128, 0, s>>>(A, lda, Bt, ldb, bias, rowScale,
                                                   C, ldc, M, K, 1, full * 4, accumulate, act);
        else
            gemm_wmma_nt<3><<<blocks, 128, 0, s>>>(A, lda, Bt, ldb, bias, rowScale,
                                                   C, ldc, M, K, 1, full * 4, accumulate, act);
    }
}

extern "C" void kernel_launch(void* const* d_in, const int* in_sizes, int n_in,
                              void* d_out, int out_size, void* d_ws, size_t ws_size,
                              hipStream_t stream)
{
    const float* enc    = (const float*)d_in[0];
    const int*   caps   = (const int*)  d_in[1];
    const int*   lens   = (const int*)  d_in[2];
    const float* emb    = (const float*)d_in[3];
    const float* W_ea   = (const float*)d_in[4];
    const float* b_ea   = (const float*)d_in[5];
    const float* W_da   = (const float*)d_in[6];
    const float* b_da   = (const float*)d_in[7];
    const float* w_fa   = (const float*)d_in[8];
    const float* b_fa   = (const float*)d_in[9];
    const float* W_ih   = (const float*)d_in[10];
    const float* b_ih   = (const float*)d_in[11];
    const float* W_hh   = (const float*)d_in[12];
    const float* b_hh   = (const float*)d_in[13];
    const float* W_h0   = (const float*)d_in[14];
    const float* b_h0   = (const float*)d_in[15];
    const float* W_c0   = (const float*)d_in[16];
    const float* b_c0   = (const float*)d_in[17];
    const float* W_beta = (const float*)d_in[18];
    const float* b_beta = (const float*)d_in[19];
    const float* W_fc   = (const float*)d_in[20];
    const float* b_fc   = (const float*)d_in[21];

    // ---- output layout (floats, concatenated in return order) ----
    float* out_f       = (float*)d_out;
    float* out_scores  = out_f;                                       // B*T*V
    float* out_caps    = out_scores + (size_t)Bsz * Tstep * Vvoc;     // B*L
    float* out_declen  = out_caps   + (size_t)Bsz * Lcap;             // B
    float* out_attn    = out_declen + Bsz;                            // B*T*P
    float* out_order   = out_attn   + (size_t)Bsz * Tstep * Ppix;     // B

    // ---- workspace carve-up ----
    char* base = (char*)d_ws;
    size_t off = 0;
    auto alloc = [&](size_t bytes) -> char* {
        off = (off + 255) & ~(size_t)255;
        char* p = base + off;
        off += bytes;
        return p;
    };
    half_t* encH   = (half_t*)alloc((size_t)Bsz * Ppix * Edim * 2);
    float*  att1   = (float*) alloc((size_t)Bsz * Ppix * Adim * 4);
    half_t* embsH  = (half_t*)alloc((size_t)Tstep * Bsz * EMBd * 2);
    half_t* WeaT   = (half_t*)alloc((size_t)Adim * Edim * 2);
    half_t* WdaT   = (half_t*)alloc((size_t)Adim * Ddim * 2);
    half_t* Wh0T   = (half_t*)alloc((size_t)Ddim * Edim * 2);
    half_t* Wc0T   = (half_t*)alloc((size_t)Ddim * Edim * 2);
    half_t* WbetaT = (half_t*)alloc((size_t)Edim * Ddim * 2);
    half_t* WfcT   = (half_t*)alloc((size_t)Vvoc * Ddim * 2);
    half_t* WihH   = (half_t*)alloc((size_t)Gdim * Xdim * 2);
    half_t* WhhH   = (half_t*)alloc((size_t)Gdim * Ddim * 2);
    half_t* meanH  = (half_t*)alloc((size_t)Bsz * Edim * 2);
    float*  hbuf   = (float*) alloc((size_t)Bsz * Ddim * 4);
    float*  cbuf   = (float*) alloc((size_t)Bsz * Ddim * 4);
    half_t* hH     = (half_t*)alloc((size_t)Bsz * Ddim * 2);
    half_t* hnewH  = (half_t*)alloc((size_t)Bsz * Ddim * 2);
    float*  att2   = (float*) alloc((size_t)Bsz * Adim * 4);
    float*  evec   = (float*) alloc((size_t)Bsz * Ppix * 4);
    float*  alpha  = (float*) alloc((size_t)Bsz * Ppix * 4);
    float*  ctx    = (float*) alloc((size_t)Bsz * Edim * 4);
    float*  glin   = (float*) alloc((size_t)Bsz * Edim * 4);
    half_t* xH     = (half_t*)alloc((size_t)Bsz * Xdim * 2);
    float*  gates  = (float*) alloc((size_t)Bsz * Gdim * 4);
    float*  amask  = (float*) alloc((size_t)Bsz * 4);
    int*    order_i  = (int*) alloc((size_t)Bsz * 4);
    int*    declen_i = (int*) alloc((size_t)Bsz * 4);
    int*    caps_i   = (int*) alloc((size_t)Bsz * Lcap * 4);
    (void)ws_size; (void)n_in; (void)in_sizes; (void)out_size;

    // ---- setup ----
    order_kernel<<<1, Bsz, 0, stream>>>(lens, order_i, declen_i, out_declen, out_order);
    caps_kernel<<<(Bsz * Lcap + 255) / 256, 256, 0, stream>>>(caps, order_i, caps_i, out_caps);
    {
        size_t n = (size_t)Bsz * Ppix * Edim;
        gather_enc_kernel<<<(unsigned)((n + 255) / 256), 256, 0, stream>>>(enc, order_i, encH);
    }
    mean_kernel<<<(Bsz * Edim + 255) / 256, 256, 0, stream>>>(enc, order_i, meanH);
    {
        size_t n = (size_t)Tstep * Bsz * EMBd;
        embs_kernel<<<(unsigned)((n + 255) / 256), 256, 0, stream>>>(caps_i, emb, embsH);
    }
    transposeW_kernel<<<(Edim * Adim + 255) / 256, 256, 0, stream>>>(W_ea, WeaT, Edim, Adim);
    transposeW_kernel<<<(Ddim * Adim + 255) / 256, 256, 0, stream>>>(W_da, WdaT, Ddim, Adim);
    transposeW_kernel<<<(Edim * Ddim + 255) / 256, 256, 0, stream>>>(W_h0, Wh0T, Edim, Ddim);
    transposeW_kernel<<<(Edim * Ddim + 255) / 256, 256, 0, stream>>>(W_c0, Wc0T, Edim, Ddim);
    transposeW_kernel<<<(Ddim * Edim + 255) / 256, 256, 0, stream>>>(W_beta, WbetaT, Ddim, Edim);
    transposeW_kernel<<<(unsigned)(((size_t)Ddim * Vvoc + 255) / 256), 256, 0, stream>>>(W_fc, WfcT, Ddim, Vvoc);
    convert_kernel<<<(unsigned)(((size_t)Gdim * Xdim + 255) / 256), 256, 0, stream>>>(W_ih, WihH, (size_t)Gdim * Xdim);
    convert_kernel<<<(unsigned)(((size_t)Gdim * Ddim + 255) / 256), 256, 0, stream>>>(W_hh, WhhH, (size_t)Gdim * Ddim);

    // att1 = encSorted @ W_ea + b_ea   (M=25088, N=512, K=1792)
    launch_gemm(stream, encH, Edim, WeaT, Edim, b_ea, nullptr, att1, Adim,
                Bsz * Ppix, Adim, Edim, 0, 0);
    // h0 = tanh(mean @ W_h0 + b_h0), c0 likewise
    launch_gemm(stream, meanH, Edim, Wh0T, Edim, b_h0, nullptr, hbuf, Ddim,
                Bsz, Ddim, Edim, 0, 1);
    launch_gemm(stream, meanH, Edim, Wc0T, Edim, b_c0, nullptr, cbuf, Ddim,
                Bsz, Ddim, Edim, 0, 1);
    f2h_kernel<<<(Bsz * Ddim + 255) / 256, 256, 0, stream>>>(hbuf, hH, Bsz * Ddim);

    // ---- time loop ----
    for (int t = 0; t < Tstep; ++t) {
        // att2 = h @ W_da + b_da
        launch_gemm(stream, hH, Ddim, WdaT, Ddim, b_da, nullptr, att2, Adim,
                    Bsz, Adim, Ddim, 0, 0);
        // e = relu(att1 + att2) . w_fa + b_fa
        attn_e_kernel<<<(Bsz * Ppix) / 8, 256, 0, stream>>>(att1, att2, w_fa, b_fa, evec);
        // alpha = softmax(e); out_attn = alpha*active
        softmax_kernel<<<Bsz, 256, 0, stream>>>(evec, alpha, out_attn, declen_i, t);
        // ctx = sum_p alpha * enc
        ctx_kernel<<<(Bsz * Edim + 255) / 256, 256, 0, stream>>>(encH, alpha, ctx);
        // gate_lin = h @ W_beta + b_beta
        launch_gemm(stream, hH, Ddim, WbetaT, Ddim, b_beta, nullptr, glin, Edim,
                    Bsz, Edim, Ddim, 0, 0);
        // x = [emb_t, sigmoid(gate_lin)*ctx]  (f16)
        buildx_kernel<<<(Bsz * Xdim + 255) / 256, 256, 0, stream>>>(
            embsH + (size_t)t * Bsz * EMBd, glin, ctx, xH);
        // gates = x @ W_ih^T + b_ih + h @ W_hh^T + b_hh
        launch_gemm(stream, xH, Xdim, WihH, Xdim, b_ih, nullptr, gates, Gdim,
                    Bsz, Gdim, Xdim, 0, 0);
        launch_gemm(stream, hH, Ddim, WhhH, Ddim, b_hh, nullptr, gates, Gdim,
                    Bsz, Gdim, Ddim, 1, 0);
        // LSTM cell + masked state update
        lstm_kernel<<<(Bsz * Ddim + 255) / 256, 256, 0, stream>>>(
            gates, hbuf, cbuf, hH, hnewH, amask, declen_i, t);
        // scores = (h_new @ W_fc + b_fc) * active  -> out[(b*T + t)*V + n]
        launch_gemm(stream, hnewH, Ddim, WfcT, Ddim, b_fc, amask,
                    out_scores + (size_t)t * Vvoc, Tstep * Vvoc,
                    Bsz, Vvoc, Ddim, 0, 0);
    }
}
